// ContrastiveLoss_41145786695939
// MI455X (gfx1250) — matrix operand order repaired and checked by reading, not compiled, not run on hardware
//
#include <hip/hip_runtime.h>
#include <hip/hip_bf16.h>
#include <math.h>

// ---- types ----------------------------------------------------------------
typedef __attribute__((ext_vector_type(16))) __bf16 v16bf;
typedef __attribute__((ext_vector_type(8)))  __bf16 v8bf;
typedef __attribute__((ext_vector_type(8)))  float  v8f;
typedef __attribute__((ext_vector_type(4)))  float  v4f;

#define DDIM 256
#define INV_T (1.0f / 0.07f)
#define LOG2E 1.4426950408889634f
#define LN2   0.6931471805599453f

// ---- CDNA5 async global->LDS helpers (ASYNCcnt path, CDNA5 ISA) --------
// VDST = 32-bit LDS byte offset (low 32 bits of generic pointer to __shared__),
// VADDR = 64-bit global address (GV mode).
__device__ __forceinline__ void async_load_b128(const void* gptr, void* lptr) {
  unsigned lds_off = (unsigned)(unsigned long long)lptr;
  asm volatile("global_load_async_to_lds_b128 %0, %1, off"
               :: "v"(lds_off), "v"(gptr) : "memory");
}
__device__ __forceinline__ void wait_async0() {
  asm volatile("s_wait_asynccnt 0" ::: "memory");
}

// ---- kernel 1: L2-normalize rows, f32 -> bf16 -----------------------------
__global__ __launch_bounds__(256) void nrm_kernel(
    const float* __restrict__ src, __bf16* __restrict__ dst, int nrows) {
  int row  = (int)((blockIdx.x * blockDim.x + threadIdx.x) >> 5);
  int lane = threadIdx.x & 31;
  if (row >= nrows) return;
  const float* p = src + (size_t)row * DDIM + lane * 8;
  v4f x0 = *(const v4f*)p;
  v4f x1 = *(const v4f*)(p + 4);
  float ss = 0.f;
#pragma unroll
  for (int i = 0; i < 4; ++i) ss += x0[i] * x0[i];
#pragma unroll
  for (int i = 0; i < 4; ++i) ss += x1[i] * x1[i];
#pragma unroll
  for (int off = 16; off > 0; off >>= 1) ss += __shfl_xor(ss, off, 32);
  float scale = 1.0f / fmaxf(sqrtf(ss), 1e-12f);
  v8bf o;
#pragma unroll
  for (int i = 0; i < 4; ++i) o[i]     = (__bf16)(x0[i] * scale);
#pragma unroll
  for (int i = 0; i < 4; ++i) o[i + 4] = (__bf16)(x1[i] * scale);
  *(v8bf*)(dst + (size_t)row * DDIM + lane * 8) = o;
}

// ---- kernel 2: fused WMMA GEMM + streaming logsumexp ----------------------
// Block = 256 threads = 8 waves; wave w owns rows [128*blk + 16*w, +16).
// Candidate tiles of 64 streamed through double-buffered LDS via
// global_load_async_to_lds_b128; compute on one buffer overlaps the async
// fill of the other. 16-byte row pad => conflict-free ds_load_b128 fragments.
#define BN 64
#define BSTRIDE 264  // bf16 elems per LDS row: 256 + 8 pad (16 bytes)

__global__ __launch_bounds__(256) void lse_kernel(
    const __bf16* __restrict__ A, const __bf16* __restrict__ C,
    float* __restrict__ lse, int M) {
  __shared__ __bf16 Bt[2][BN * BSTRIDE];  // 2 x 33,792 bytes

  int t    = threadIdx.x;
  int wave = t >> 5;
  int lane = t & 31;
  int half = lane >> 4;   // 0 or 1
  int l15  = lane & 15;
  int rowBase = blockIdx.x * 128 + wave * 16;

  // per-thread chunk assignment for tile fills: 2048 16B chunks / 256 threads
  // chunk j: row = (j*256+t)>>5, 16B offset within row = ((j*256+t)&31)*16
  int brow0  = t >> 5;        // rows t/32, advancing by 8 per j
  int bchnk  = (t & 31) * 8;  // bf16 element offset of this thread's chunk

  // kick off async fill of tile 0 into buffer 0
#pragma unroll
  for (int j = 0; j < 8; ++j) {
    int brow = brow0 + j * 8;
    async_load_b128(C + (size_t)brow * DDIM + bchnk,
                    &Bt[0][brow * BSTRIDE + bchnk]);
  }

  // --- preload A fragments (16x32 bf16 per K-step, ISA lane layout) ---
  v16bf afrag[8];
  {
    const __bf16* arow = A + (size_t)(rowBase + l15) * DDIM;
#pragma unroll
    for (int ks = 0; ks < 8; ++ks) {
      int k0 = ks * 32 + half * 8;
      v8bf lo = *(const v8bf*)(arow + k0);
      v8bf hi = *(const v8bf*)(arow + k0 + 16);
      afrag[ks] = __builtin_shufflevector(lo, hi, 0, 1, 2, 3, 4, 5, 6, 7,
                                          8, 9, 10, 11, 12, 13, 14, 15);
    }
  }

  // per-lane exp-sum state: reg v of acc -> row 8*half + v
  float s[8];
#pragma unroll
  for (int v = 0; v < 8; ++v) s[v] = 0.f;

  const float scale = INV_T * LOG2E;  // logits bounded => no running max needed
  const int nTiles = M / BN;

  for (int it = 0; it < nTiles; ++it) {
    wait_async0();      // my async loads for tile `it` have landed in LDS
    __syncthreads();    // everyone's landed; everyone done reading buf (it-1)&1

    if (it + 1 < nTiles) {  // overlap: fill the other buffer during compute
      const __bf16* cnxt = C + (size_t)(it + 1) * BN * DDIM;
      __bf16* bnxt = &Bt[(it + 1) & 1][0];
#pragma unroll
      for (int j = 0; j < 8; ++j) {
        int brow = brow0 + j * 8;
        async_load_b128(cnxt + (size_t)brow * DDIM + bchnk,
                        bnxt + brow * BSTRIDE + bchnk);
      }
    }

    const __bf16* bcur = &Bt[it & 1][0];
    v8f acc[4] = {};  // 4 column sub-tiles of 16
#pragma unroll
    for (int ks = 0; ks < 8; ++ks) {
#pragma unroll
      for (int n = 0; n < 4; ++n) {
        const __bf16* bp = bcur + (n * 16 + l15) * BSTRIDE + ks * 32 + half * 8;
        v8bf lo = *(const v8bf*)bp;
        v8bf hi = *(const v8bf*)(bp + 16);
        v16bf bfrag = __builtin_shufflevector(lo, hi, 0, 1, 2, 3, 4, 5, 6, 7,
                                              8, 9, 10, 11, 12, 13, 14, 15);
        acc[n] = __builtin_amdgcn_wmma_f32_16x16x32_bf16(
            false, afrag[ks], false, bfrag, (short)0, acc[n], false, false);
      }
    }

#pragma unroll
    for (int n = 0; n < 4; ++n)
#pragma unroll
      for (int v = 0; v < 8; ++v) s[v] += exp2f(acc[n][v] * scale);
  }

  // merge the 16 lanes of each half (they hold the same 8 rows)
#pragma unroll
  for (int v = 0; v < 8; ++v) {
    float sv = s[v];
#pragma unroll
    for (int off = 1; off < 16; off <<= 1) sv += __shfl_xor(sv, off, 32);
    if (l15 == 0) lse[rowBase + half * 8 + v] = log2f(sv) * LN2;
  }
}

// ---- kernel 3: per-row NLL = lse - <a_i, c_target>/T ----------------------
__global__ __launch_bounds__(256) void nll_kernel(
    const __bf16* __restrict__ A, const __bf16* __restrict__ C,
    const float* __restrict__ lse, const int* __restrict__ tgt,
    float* __restrict__ nll, int N) {
  int row  = (int)((blockIdx.x * blockDim.x + threadIdx.x) >> 5);
  int lane = threadIdx.x & 31;
  if (row >= N) return;
  int tg = tgt[row];
  v8bf av = *(const v8bf*)(A + (size_t)row * DDIM + lane * 8);
  v8bf cv = *(const v8bf*)(C + (size_t)tg  * DDIM + lane * 8);
  float acc = 0.f;
#pragma unroll
  for (int i = 0; i < 8; ++i) acc += (float)av[i] * (float)cv[i];
#pragma unroll
  for (int off = 16; off > 0; off >>= 1) acc += __shfl_xor(acc, off, 32);
  if (lane == 0) nll[row] = lse[row] - acc * INV_T;
}

// ---- kernel 4: deterministic mean reduce ----------------------------------
__global__ __launch_bounds__(256) void reduce_kernel(
    const float* __restrict__ nll, float* __restrict__ out, int N) {
  __shared__ float sm[256];
  float s = 0.f;
  for (int i = threadIdx.x; i < N; i += 256) s += nll[i];
  sm[threadIdx.x] = s;
  __syncthreads();
  for (int st = 128; st > 0; st >>= 1) {
    if ((int)threadIdx.x < st) sm[threadIdx.x] += sm[threadIdx.x + st];
    __syncthreads();
  }
  if (threadIdx.x == 0) out[0] = sm[0] / (float)N;
}

// ---- launch ---------------------------------------------------------------
extern "C" void kernel_launch(void* const* d_in, const int* in_sizes, int n_in,
                              void* d_out, int out_size, void* d_ws, size_t ws_size,
                              hipStream_t stream) {
  const float* anchors    = (const float*)d_in[0];
  const float* candidates = (const float*)d_in[1];
  const int*   targets    = (const int*)d_in[2];
  float*       out        = (float*)d_out;

  const int N = in_sizes[0] / DDIM;  // 16384
  const int M = in_sizes[1] / DDIM;  // 16384

  char* ws = (char*)d_ws;
  __bf16* An  = (__bf16*)ws;                                  // N*256 bf16 (8 MB)
  __bf16* Cn  = (__bf16*)(ws + (size_t)N * DDIM * 2);         // M*256 bf16 (8 MB)
  float*  lse = (float*)(ws + (size_t)(N + M) * DDIM * 2);    // N f32
  float*  nll = lse + N;                                      // N f32

  nrm_kernel<<<(N + 7) / 8, 256, 0, stream>>>(anchors, An, N);
  nrm_kernel<<<(M + 7) / 8, 256, 0, stream>>>(candidates, Cn, M);
  lse_kernel<<<N / 128, 256, 0, stream>>>(An, Cn, lse, M);
  nll_kernel<<<(N + 7) / 8, 256, 0, stream>>>(An, Cn, lse, targets, nll, N);
  reduce_kernel<<<1, 256, 0, stream>>>(nll, out, N);
}
